// Block_8735963480467
// MI455X (gfx1250) — matrix operand order repaired and checked by reading, not compiled, not run on hardware
//
#include <hip/hip_runtime.h>
#include <hip/hip_bf16.h>

// ---------------------------------------------------------------------------
// Types matching the probe-confirmed WMMA builtin signatures
// ---------------------------------------------------------------------------
typedef __attribute__((ext_vector_type(16))) __bf16 v16bf;
typedef __attribute__((ext_vector_type(8)))  __bf16 bf16x8;
typedef __attribute__((ext_vector_type(8)))  float  v8f;

#define THREADS 256

// ---------------------------------------------------------------------------
// CDNA5 async copy helpers (ASYNCcnt-tracked LDS DMA, ISA ch.10 async ops).
// Inline asm so it works on both toolchains regardless of builtin arity.
//   global_load_async_to_lds_b128 vDstLds, v[gaddr:gaddr+1], off
// LDS dest address is the per-lane VGPR value (low 32 bits of the generic
// pointer == LDS byte address).
// ---------------------------------------------------------------------------
__device__ __forceinline__ void async_load_b128(const void* gaddr, void* ldsaddr) {
    unsigned           ld = (unsigned)(unsigned long long)ldsaddr;
    unsigned long long ga = (unsigned long long)gaddr;
    asm volatile("global_load_async_to_lds_b128 %0, %1, off"
                 :: "v"(ld), "v"(ga) : "memory");
}
__device__ __forceinline__ void wait_async0() {
    asm volatile("s_wait_asynccnt 0x0" ::: "memory");
}

// ---------------------------------------------------------------------------
// Tiled transpose + f32->bf16 convert:  in[R][C] (f32) -> out[C][R] (bf16)
// ---------------------------------------------------------------------------
__global__ void transpose_f32_bf16_kernel(const float* __restrict__ in,
                                          __bf16* __restrict__ out,
                                          int R, int C) {
    __shared__ float tile[32][33];
    const int c0 = blockIdx.x * 32;
    const int r0 = blockIdx.y * 32;
    const int tx = threadIdx.x, ty = threadIdx.y;
#pragma unroll
    for (int i = 0; i < 4; ++i)
        tile[ty + 8 * i][tx] = in[(long)(r0 + ty + 8 * i) * C + c0 + tx];
    __syncthreads();
#pragma unroll
    for (int i = 0; i < 4; ++i)
        out[(long)(c0 + ty + 8 * i) * R + r0 + tx] =
            (__bf16)tile[tx][ty + 8 * i];
}

// ---------------------------------------------------------------------------
// Per-(b,h) transpose of the V slice of qkv (bf16 -> bf16):
//   vT[bh][d][s] = qkv[(b*S+s)*3072 + 2048 + h*64 + d]
// ---------------------------------------------------------------------------
__global__ void transpose_v_kernel(const __bf16* __restrict__ qkv,
                                   __bf16* __restrict__ vT) {
    __shared__ __bf16 tile[32][33];
    const int bh = blockIdx.z;
    const int b  = bh >> 4;
    const int h  = bh & 15;
    const __bf16* in = qkv + ((long)b * 2048) * 3072 + 2048 + h * 64;
    __bf16* out = vT + (long)bh * 64 * 2048;
    const int d0 = blockIdx.x * 32;
    const int s0 = blockIdx.y * 32;
    const int tx = threadIdx.x, ty = threadIdx.y;
#pragma unroll
    for (int i = 0; i < 4; ++i)
        tile[ty + 8 * i][tx] = in[(long)(s0 + ty + 8 * i) * 3072 + d0 + tx];
    __syncthreads();
#pragma unroll
    for (int i = 0; i < 4; ++i)
        out[(long)(d0 + ty + 8 * i) * 2048 + s0 + tx] = tile[tx][ty + 8 * i];
}

// ---------------------------------------------------------------------------
// Row LayerNorm: f32 in -> bf16 out. One block per row.
// ---------------------------------------------------------------------------
__global__ void ln_rows_kernel(const float* __restrict__ X,
                               const float* __restrict__ w,
                               const float* __restrict__ b,
                               __bf16* __restrict__ Y, int cols) {
    const int row = blockIdx.x;
    const int tid = threadIdx.x;
    const float* x = X + (long)row * cols;
    float s = 0.f, s2 = 0.f;
    for (int c = tid; c < cols; c += THREADS) {
        float v = x[c];
        s += v; s2 += v * v;
    }
    __shared__ float rs[THREADS], rs2[THREADS];
    rs[tid] = s; rs2[tid] = s2;
    __syncthreads();
    for (int off = THREADS / 2; off > 0; off >>= 1) {
        if (tid < off) { rs[tid] += rs[tid + off]; rs2[tid] += rs2[tid + off]; }
        __syncthreads();
    }
    float mu  = rs[0] / (float)cols;
    float var = rs2[0] / (float)cols - mu * mu;
    float inv = rsqrtf(var + 1e-5f);
    __bf16* y = Y + (long)row * cols;
    for (int c = tid; c < cols; c += THREADS)
        y[c] = (__bf16)(((x[c] - mu) * inv) * w[c] + b[c]);
}

// ---------------------------------------------------------------------------
// Row softmax over all (bh, s_q) rows: f32 scores -> bf16 probabilities.
// ---------------------------------------------------------------------------
__global__ void softmax_rows_kernel(const float* __restrict__ S,
                                    __bf16* __restrict__ P, int cols) {
    const long row = blockIdx.x;
    const int  tid = threadIdx.x;
    const float* s = S + row * cols;
    __shared__ float red[THREADS];

    float mx = -3.0e38f;
    for (int c = tid; c < cols; c += THREADS) mx = fmaxf(mx, s[c]);
    red[tid] = mx;
    __syncthreads();
    for (int off = THREADS / 2; off > 0; off >>= 1) {
        if (tid < off) red[tid] = fmaxf(red[tid], red[tid + off]);
        __syncthreads();
    }
    mx = red[0];
    __syncthreads();

    float sum = 0.f;
    for (int c = tid; c < cols; c += THREADS) sum += expf(s[c] - mx);
    red[tid] = sum;
    __syncthreads();
    for (int off = THREADS / 2; off > 0; off >>= 1) {
        if (tid < off) red[tid] += red[tid + off];
        __syncthreads();
    }
    float inv = 1.0f / red[0];

    __bf16* p = P + row * cols;
    for (int c = tid; c < cols; c += THREADS)
        p[c] = (__bf16)(expf(s[c] - mx) * inv);
}

// ---------------------------------------------------------------------------
// WMMA bf16 GEMM (optionally batched over z = (b,h)):
//   C[z] = act(alpha * A[z] @ Bt[z]^T + bias) [+ res]
//   A : MxK bf16 row-major, leading dim lda
//   Bt: NxK bf16 row-major ("N-major" B), leading dim ldb
//   C : f32 or bf16, leading dim ldc
//   z-offsets (elements): off = (z/16)*z?b + (z%16)*z?h for A, Bt, C.
//
// Block = 256 threads = 8 waves; each wave owns 64x32 output (4x2 tiles of
// 16x16, 8 v_wmma per K-step). BK = 32. LDS is double-buffered and filled by
// the CDNA5 async-copy engine (global_load_async_to_lds_b128): next K-tile
// streams into buf^1 while WMMAs consume buf, then s_wait_asynccnt 0 +
// barrier swap. No VGPR staging, no ds_store traffic.
// ---------------------------------------------------------------------------
template <int WM, int WN, bool OUTBF16, bool DO_GELU, bool HAS_BIAS, bool HAS_RES>
__global__ __launch_bounds__(THREADS, 2)
void gemm_wmma_kernel(const __bf16* __restrict__ A,
                      const __bf16* __restrict__ Bt,
                      void* __restrict__ Cv,
                      const float* __restrict__ bias,
                      const float* __restrict__ res,
                      int K, long lda, long ldb, long ldc, float alpha,
                      long zAb, long zAh, long zBb, long zBh,
                      long zCb, long zCh) {
    constexpr int BM = WM * 64;
    constexpr int BN = WN * 32;
    constexpr int BK = 32;
    constexpr int LS = BK + 8;          // 40 bf16 = 80 B, 16 B aligned

    __shared__ __align__(16) __bf16 ldsA[2][BM * LS];   // row-major [BM][BK]
    __shared__ __align__(16) __bf16 ldsB[2][BN * LS];   // N-major   [BN][BK]

    const int z  = blockIdx.z;
    const int zb = z >> 4, zh = z & 15;
    A  += (long)zb * zAb + (long)zh * zAh;
    Bt += (long)zb * zBb + (long)zh * zBh;
    const long cOff = (long)zb * zCb + (long)zh * zCh;

    const int tid   = threadIdx.x;
    const int lane  = tid & 31;
    const int wave  = tid >> 5;
    const int waveM = wave / WN;
    const int waveN = wave % WN;
    const long rowBlock = (long)blockIdx.y * BM;
    const long colBlock = (long)blockIdx.x * BN;

    v8f c[4][2];
#pragma unroll
    for (int i = 0; i < 4; ++i)
#pragma unroll
        for (int j = 0; j < 2; ++j) {
            v8f zz = {0.f, 0.f, 0.f, 0.f, 0.f, 0.f, 0.f, 0.f};
            c[i][j] = zz;
        }

    constexpr int ACH = (BM * BK / 8) / THREADS;  // 8-elem A chunks / thread
    constexpr int BCH = (BN * BK / 8) / THREADS;  // 8-elem B chunks / thread

    auto issue_tile = [&](int buf, int k0) {
#pragma unroll
        for (int it = 0; it < ACH; ++it) {
            int chunk = tid + it * THREADS;
            int row = chunk >> 2, kc = chunk & 3;
            async_load_b128(A + (rowBlock + row) * lda + k0 + kc * 8,
                            &ldsA[buf][row * LS + kc * 8]);
        }
#pragma unroll
        for (int it = 0; it < BCH; ++it) {
            int chunk = tid + it * THREADS;
            int n = chunk >> 2, kc = chunk & 3;
            async_load_b128(Bt + (colBlock + n) * ldb + k0 + kc * 8,
                            &ldsB[buf][n * LS + kc * 8]);
        }
    };

    // prologue: fill buffer 0
    issue_tile(0, 0);
    wait_async0();
    __syncthreads();

    const int hf  = lane >> 4;
    const int l15 = lane & 15;
    int cur = 0;

    for (int k0 = 0; k0 < K; k0 += BK) {
        // stream next K-tile into the other buffer while WMMAs run
        if (k0 + BK < K) issue_tile(cur ^ 1, k0 + BK);

        v16bf aF[4], bF[2];
#pragma unroll
        for (int i = 0; i < 4; ++i) {
            int m = waveM * 64 + i * 16 + l15;
            bf16x8 lo = *(const bf16x8*)&ldsA[cur][m * LS + hf * 8];
            bf16x8 hi = *(const bf16x8*)&ldsA[cur][m * LS + 16 + hf * 8];
            ((bf16x8*)&aF[i])[0] = lo;
            ((bf16x8*)&aF[i])[1] = hi;
        }
#pragma unroll
        for (int j = 0; j < 2; ++j) {
            int n = waveN * 32 + j * 16 + l15;
            const bf16x8* p = (const bf16x8*)&ldsB[cur][n * LS + hf * 8];
            ((bf16x8*)&bF[j])[0] = p[0];
            ((bf16x8*)&bF[j])[1] = p[2];
        }
#pragma unroll
        for (int i = 0; i < 4; ++i)
#pragma unroll
            for (int j = 0; j < 2; ++j)
                c[i][j] = __builtin_amdgcn_wmma_f32_16x16x32_bf16(
                    false, aF[i], false, bF[j], (short)0, c[i][j],
                    false, false);

        // next buffer complete + all waves done reading before swap
        wait_async0();
        __syncthreads();
        cur ^= 1;
    }

    // ---- epilogue: alpha, bias, GELU, residual, store ----
#pragma unroll
    for (int j = 0; j < 2; ++j) {
        long gc = colBlock + waveN * 32 + j * 16 + (lane & 15);
        float bv = HAS_BIAS ? bias[gc] : 0.0f;
#pragma unroll
        for (int i = 0; i < 4; ++i) {
            long gr0 = rowBlock + waveM * 64 + i * 16 + (lane >> 4) * 8;
#pragma unroll
            for (int r = 0; r < 8; ++r) {
                long gr = gr0 + r;
                float v = c[i][j][r] * alpha + bv;
                if (DO_GELU) v = 0.5f * v * (1.0f + erff(v * 0.70710678f));
                if (HAS_RES) v += res[gr * ldc + gc];
                if (OUTBF16) ((__bf16*)Cv)[cOff + gr * ldc + gc] = (__bf16)v;
                else         ((float*)Cv)[cOff + gr * ldc + gc] = v;
            }
        }
    }
}

// ---------------------------------------------------------------------------
// Host-side orchestration
// ---------------------------------------------------------------------------
extern "C" void kernel_launch(void* const* d_in, const int* in_sizes, int n_in,
                              void* d_out, int out_size, void* d_ws, size_t ws_size,
                              hipStream_t stream) {
    (void)in_sizes; (void)n_in; (void)out_size; (void)ws_size;

    constexpr int  Bsz = 4, S = 2048, DIM = 1024, HEADS = 16, DH = 64, MLP = 4096;
    constexpr long ROWS = (long)Bsz * S;          // 8192
    constexpr long QKVW = 3 * DIM;                // 3072
    constexpr long BH   = (long)Bsz * HEADS;      // 64

    const float* x     = (const float*)d_in[0];
    const float* ln1_w = (const float*)d_in[1];
    const float* ln1_b = (const float*)d_in[2];
    const float* w_qkv = (const float*)d_in[3];
    const float* w_out = (const float*)d_in[4];
    const float* b_out = (const float*)d_in[5];
    const float* ln2_w = (const float*)d_in[6];
    const float* ln2_b = (const float*)d_in[7];
    const float* w1    = (const float*)d_in[8];
    const float* b1    = (const float*)d_in[9];
    const float* w2    = (const float*)d_in[10];
    const float* b2    = (const float*)d_in[11];
    float* out = (float*)d_out;

    // Workspace carve-up (bytes)
    char* ws = (char*)d_ws;
    __bf16* wqkvT  = (__bf16*)(ws + 0);                    //   6 MB [3072][1024]
    __bf16* woutT  = (__bf16*)(ws + 6291456);              //   2 MB [1024][1024]
    __bf16* w1T    = (__bf16*)(ws + 8388608);              //   8 MB [4096][1024]
    __bf16* w2T    = (__bf16*)(ws + 16777216);             //   8 MB [1024][4096]
    __bf16* h1     = (__bf16*)(ws + 25165824);             //  16 MB
    __bf16* qkv    = (__bf16*)(ws + 41943040);             //  48 MB
    __bf16* vT     = (__bf16*)(ws + 92274688);             //  16 MB [bh][64][2048]
    __bf16* obuf   = (__bf16*)(ws + 109051904);            //  16 MB
    float*  x1     = (float*) (ws + 125829120);            //  32 MB
    __bf16* h2     = (__bf16*)(ws + 159383552);            //  16 MB
    __bf16* m1     = (__bf16*)(ws + 175112192);            //  64 MB
    float*  scores = (float*) (ws + 239075328);            //   1 GB [bh][S][S]
    __bf16* probs  = (__bf16*)(ws + 1312817152);           // 512 MB [bh][S][S]

    const dim3 tb(32, 8);
    const long Z0 = 0;

    // 1) Weights -> bf16, transposed to N-major
    transpose_f32_bf16_kernel<<<dim3(QKVW / 32, DIM / 32), tb, 0, stream>>>(w_qkv, wqkvT, DIM, QKVW);
    transpose_f32_bf16_kernel<<<dim3(DIM / 32, DIM / 32),  tb, 0, stream>>>(w_out, woutT, DIM, DIM);
    transpose_f32_bf16_kernel<<<dim3(MLP / 32, DIM / 32),  tb, 0, stream>>>(w1,    w1T,   DIM, MLP);
    transpose_f32_bf16_kernel<<<dim3(DIM / 32, MLP / 32),  tb, 0, stream>>>(w2,    w2T,   MLP, DIM);

    // 2) LN1: x -> h1 (bf16)
    ln_rows_kernel<<<ROWS, THREADS, 0, stream>>>(x, ln1_w, ln1_b, h1, DIM);

    // 3) QKV projection: qkv = h1 @ w_qkv   (8192 x 3072 x 1024), bf16 out
    gemm_wmma_kernel<2, 4, true, false, false, false>
        <<<dim3(QKVW / 128, ROWS / 128, 1), THREADS, 0, stream>>>(
            h1, wqkvT, qkv, nullptr, nullptr, DIM, DIM, DIM, QKVW, 1.0f,
            Z0, Z0, Z0, Z0, Z0, Z0);

    // 4) Transpose V slices: vT[bh][d][s]
    transpose_v_kernel<<<dim3(DH / 32, S / 32, BH), tb, 0, stream>>>(qkv, vT);

    // 5a) Batched scores = scale * Q @ K^T over all 64 heads, f32 out
    gemm_wmma_kernel<2, 4, false, false, false, false>
        <<<dim3(S / 128, S / 128, BH), THREADS, 0, stream>>>(
            qkv /*Q base*/, qkv + DIM /*K base*/, scores, nullptr, nullptr,
            DH, QKVW, QKVW, S, 0.125f,
            /*zA*/ (long)S * QKVW, DH, /*zB*/ (long)S * QKVW, DH,
            /*zC*/ 16L * S * S, (long)S * S);

    // 5b) Batched softmax over all bh*S rows, bf16 probs
    softmax_rows_kernel<<<BH * S, THREADS, 0, stream>>>(scores, probs, S);

    // 5c) Batched o = probs @ V, bf16 out into [b][s][h*64+d]
    gemm_wmma_kernel<4, 2, true, false, false, false>
        <<<dim3(DH / 64, S / 256, BH), THREADS, 0, stream>>>(
            probs, vT, obuf, nullptr, nullptr, S, S, S, DIM, 1.0f,
            /*zA*/ 16L * S * S, (long)S * S, /*zB*/ 16L * DH * S, (long)DH * S,
            /*zC*/ (long)S * DIM, DH);

    // 6) Out projection + residual: x1 = obuf @ w_out + b_out + x, f32 out
    gemm_wmma_kernel<2, 4, false, false, true, true>
        <<<dim3(DIM / 128, ROWS / 128, 1), THREADS, 0, stream>>>(
            obuf, woutT, x1, b_out, x, DIM, DIM, DIM, DIM, 1.0f,
            Z0, Z0, Z0, Z0, Z0, Z0);

    // 7) LN2: x1 -> h2 (bf16)
    ln_rows_kernel<<<ROWS, THREADS, 0, stream>>>(x1, ln2_w, ln2_b, h2, DIM);

    // 8) MLP up: m1 = gelu(h2 @ w1 + b1), bf16 out  (8192 x 4096 x 1024)
    gemm_wmma_kernel<2, 4, true, true, true, false>
        <<<dim3(MLP / 128, ROWS / 128, 1), THREADS, 0, stream>>>(
            h2, w1T, m1, b1, nullptr, DIM, DIM, DIM, MLP, 1.0f,
            Z0, Z0, Z0, Z0, Z0, Z0);

    // 9) MLP down: out = gelu(m1 @ w2 + b2) + x1, f32 out  (8192 x 1024 x 4096)
    gemm_wmma_kernel<2, 4, false, true, true, true>
        <<<dim3(DIM / 128, ROWS / 128, 1), THREADS, 0, stream>>>(
            m1, w2T, out, b2, x1, MLP, MLP, MLP, DIM, 1.0f,
            Z0, Z0, Z0, Z0, Z0, Z0);
}
